// Evolution_67645734912888
// MI455X (gfx1250) — matrix operand order repaired and checked by reading, not compiled, not run on hardware
//
#include <hip/hip_runtime.h>
#include <hip/hip_bf16.h>
#include <cstdint>
#include <cstddef>

// ---------------------------------------------------------------------------
// Types for CDNA5 WMMA (wave32): D(16x16 f32) = A(16x32 f16) * B(32x16 f16) + C
// ---------------------------------------------------------------------------
typedef __attribute__((ext_vector_type(16))) _Float16 v16h;
typedef __attribute__((ext_vector_type(8)))  _Float16 v8h;
typedef __attribute__((ext_vector_type(8)))  float    v8f;
typedef __attribute__((ext_vector_type(4)))  int      v4i;

union H16 { v16h v; v8h h[2]; };

__device__ __forceinline__ v8f wmma_f16(v16h a, v16h b, v8f c) {
  // 8 args: (neg_a, A, neg_b, B, c_mod, C, reuse_a, reuse_b)
  return __builtin_amdgcn_wmma_f32_16x16x32_f16(false, a, false, b, (short)0, c,
                                                false, false);
}

#if __has_builtin(__builtin_amdgcn_global_load_async_to_lds_b128)
#define HAVE_ASYNC_LDS 1
#else
#define HAVE_ASYNC_LDS 0
#endif

#if HAVE_ASYNC_LDS
__device__ __forceinline__ void async_copy_b128(const void* gsrc, void* ldst) {
  // global -> LDS with no VGPR staging; tracked on ASYNCcnt.
  // Builtin params are int4 pointers in AS1 (global) / AS3 (LDS); cast through
  // integers, HIP-header style. Flat LDS address truncates to AS3 offset.
  __builtin_amdgcn_global_load_async_to_lds_b128(
      (__attribute__((address_space(1))) v4i*)(uintptr_t)gsrc,
      (__attribute__((address_space(3))) v4i*)(uint32_t)(uintptr_t)ldst,
      0, 0);
}
__device__ __forceinline__ void wait_async0() {
#if __has_builtin(__builtin_amdgcn_s_wait_asynccnt)
  __builtin_amdgcn_s_wait_asynccnt(0);
#else
  asm volatile("s_wait_asynccnt 0x0" ::: "memory");
#endif
}
#endif

#define NPOLY 512
#define PPTS  128
#define SD    128     // snake hidden dim
#define FD    67      // 64 feat + 2 coord + 1 cls
#define FDPAD 96      // pad to multiple of 32 for K-blocking
#define KTAP  9
#define CCH   64
#define HH    128
#define WW    128
#define STCH  1024    // 8 * SD concatenated states
#define FUS   256
#define HD1   256
#define HD2   64

// ---------------------------------------------------------------------------
// Weight conversion kernels (f32 -> f16, WMMA-friendly layouts)
// ---------------------------------------------------------------------------
__global__ void k_cvt_head(const float* __restrict__ src, _Float16* __restrict__ dst) {
  int i = blockIdx.x * blockDim.x + threadIdx.x;          // [k][oc][icpad]
  if (i >= KTAP * SD * FDPAD) return;
  int ic = i % FDPAD; int t = i / FDPAD; int oc = t % SD; int k = t / SD;
  float v = (ic < FD) ? src[(oc * FD + ic) * KTAP + k] : 0.0f;
  dst[i] = (_Float16)v;
}

__global__ void k_cvt_res(const float* __restrict__ src, _Float16* __restrict__ dst) {
  int i = blockIdx.x * blockDim.x + threadIdx.x;          // [l][k][oc][ic]
  if (i >= 7 * KTAP * SD * SD) return;
  int ic = i & 127; int t = i >> 7; int oc = t & 127; t >>= 7;
  int k = t % KTAP; int l = t / KTAP;
  dst[i] = (_Float16)src[(((l * SD + oc) * SD + ic) * KTAP) + k];
}

__global__ void k_cvt_copy(const float* __restrict__ src, _Float16* __restrict__ dst, int n) {
  for (int i = blockIdx.x * blockDim.x + threadIdx.x; i < n; i += gridDim.x * blockDim.x)
    dst[i] = (_Float16)src[i];
}

// ---------------------------------------------------------------------------
// Bilinear gather + input assembly: xpad[n][p][0..95] f16 (point-major)
// ---------------------------------------------------------------------------
__global__ __launch_bounds__(128)
void k_gather(const float* __restrict__ cnn, const float* __restrict__ ipoly,
              const float* __restrict__ cpoly, const float* __restrict__ itcls,
              const int* __restrict__ ind, _Float16* __restrict__ xpad) {
  int q = blockIdx.x * blockDim.x + threadIdx.x;
  if (q >= NPOLY * PPTS) return;
  int n = q >> 7;
  float x = ipoly[q * 2 + 0] - 0.5f;
  float y = ipoly[q * 2 + 1] - 0.5f;
  float x0f = floorf(x), y0f = floorf(y);
  float wx = x - x0f, wy = y - y0f;
  int x0 = (int)x0f, y0 = (int)y0f;
  int b = ind[n];
  float w00 = (1.f - wx) * (1.f - wy), w01 = wx * (1.f - wy);
  float w10 = (1.f - wx) * wy,         w11 = wx * wy;
  auto ok = [](int yy, int xx) { return (xx >= 0) && (xx < WW) && (yy >= 0) && (yy < HH); };
  float m00 = ok(y0, x0)         ? w00 : 0.f;
  float m01 = ok(y0, x0 + 1)     ? w01 : 0.f;
  float m10 = ok(y0 + 1, x0)     ? w10 : 0.f;
  float m11 = ok(y0 + 1, x0 + 1) ? w11 : 0.f;
  int xc0 = min(max(x0, 0), WW - 1),     xc1 = min(max(x0 + 1, 0), WW - 1);
  int yc0 = min(max(y0, 0), HH - 1),     yc1 = min(max(y0 + 1, 0), HH - 1);
  const float* base = cnn + (size_t)b * CCH * HH * WW;
  size_t o00 = (size_t)yc0 * WW + xc0, o01 = (size_t)yc0 * WW + xc1;
  size_t o10 = (size_t)yc1 * WW + xc0, o11 = (size_t)yc1 * WW + xc1;
  _Float16* dst = xpad + (size_t)q * FDPAD;
  for (int c = 0; c < CCH; ++c) {
    const float* pc = base + (size_t)c * HH * WW;
    float v = pc[o00] * m00 + pc[o01] * m01 + pc[o10] * m10 + pc[o11] * m11;
    dst[c] = (_Float16)v;
  }
  dst[64] = (_Float16)(cpoly[q * 2 + 0] * 4.0f);
  dst[65] = (_Float16)(cpoly[q * 2 + 1] * 4.0f);
  dst[66] = (_Float16)itcls[q];
  for (int c = FD; c < FDPAD; ++c) dst[c] = (_Float16)0.0f;
}

// ---------------------------------------------------------------------------
// Circular dilated conv as 9 shifted WMMA GEMMs. One block per polygon.
//   xin : point-major input rows (stride in_stride halves), Cin = 96 or 128
//   out : state buffer slice base (row stride STCH halves)
// LDS tile fill uses the CDNA5 async global->LDS data movers when available.
// ---------------------------------------------------------------------------
__global__ __launch_bounds__(256)
void k_conv(const _Float16* __restrict__ xin, int in_stride, int Cin,
            const _Float16* __restrict__ wgt, const float* __restrict__ bias,
            _Float16* __restrict__ out, int dil, int do_res) {
  __shared__ __align__(16) _Float16 lds[PPTS * SD];   // 32 KB max
  int n = blockIdx.x, tid = threadIdx.x;
  const _Float16* xrow0 = xin + (size_t)n * PPTS * in_stride;
  int chunks = Cin >> 3;
#if HAVE_ASYNC_LDS
  for (int i = tid; i < PPTS * chunks; i += 256) {
    int p = i / chunks, j = i - p * chunks;
    async_copy_b128(&xrow0[(size_t)p * in_stride + j * 8], &lds[p * Cin + j * 8]);
  }
  wait_async0();
#else
  for (int i = tid; i < PPTS * chunks; i += 256) {
    int p = i / chunks, j = i - p * chunks;
    *(v8h*)&lds[p * Cin + j * 8] = *(const v8h*)&xrow0[(size_t)p * in_stride + j * 8];
  }
#endif
  __syncthreads();

  int wave = tid >> 5, lane = tid & 31, ln = lane & 15, hi = lane >> 4;
  int m0 = wave * 16;                                   // 8 waves * 16 rows = 128 out ch
  float bv[8];
#pragma unroll
  for (int r = 0; r < 8; ++r) bv[r] = bias[m0 + hi * 8 + r];
  v8f acc[8];
#pragma unroll
  for (int t = 0; t < 8; ++t)
#pragma unroll
    for (int r = 0; r < 8; ++r) acc[t][r] = bv[r];

  int cb = Cin >> 5;
  for (int k = 0; k < KTAP; ++k) {
    int shift = (k - 4) * dil;
    for (int c = 0; c < cb; ++c) {
      const _Float16* wrow = wgt + ((size_t)(k * SD + m0 + ln) * Cin) + c * 32;
      __builtin_prefetch(wrow + Cin, 0, 1);             // global_prefetch_b8
      H16 A;
      A.h[0] = *(const v8h*)(wrow + hi * 16);
      A.h[1] = *(const v8h*)(wrow + hi * 16 + 8);
#pragma unroll
      for (int t = 0; t < 8; ++t) {
        int p = (t * 16 + ln + shift) & 127;            // circular shift in LDS
        const _Float16* br = &lds[p * Cin + c * 32 + hi * 16];
        H16 B;
        B.h[0] = *(const v8h*)br;
        B.h[1] = *(const v8h*)(br + 8);
        acc[t] = wmma_f16(A.v, B.v, acc[t]);
      }
    }
  }

#pragma unroll
  for (int t = 0; t < 8; ++t) {
    int p = t * 16 + ln;
    v8h rv;
    if (do_res) rv = *(const v8h*)&lds[p * Cin + m0 + hi * 8];
    v8h res;
#pragma unroll
    for (int r = 0; r < 8; ++r) {
      float v = acc[t][r];
      v = v > 0.f ? v : 0.f;                            // relu
      if (do_res) v += (float)rv[r];                    // residual
      res[r] = (_Float16)v;
    }
    *(v8h*)&out[(size_t)(n * PPTS + p) * STCH + m0 + hi * 8] = res;
  }
}

// ---------------------------------------------------------------------------
// Fusion 1x1 (1024->256) + max over P -> g[n][256]
// ---------------------------------------------------------------------------
__global__ __launch_bounds__(256)
void k_fusion(const _Float16* __restrict__ state, const _Float16* __restrict__ wf,
              const float* __restrict__ bf, float* __restrict__ g) {
  int n = blockIdx.x, tid = threadIdx.x;
  int wave = tid >> 5, lane = tid & 31, ln = lane & 15, hi = lane >> 4;
  int m0 = wave * 32;                                   // 8 waves * 32 rows = 256
  float b0[8], b1[8], rmax0[8], rmax1[8];
#pragma unroll
  for (int r = 0; r < 8; ++r) {
    b0[r] = bf[m0 + hi * 8 + r];
    b1[r] = bf[m0 + 16 + hi * 8 + r];
    rmax0[r] = -3.4e38f; rmax1[r] = -3.4e38f;
  }
  for (int t = 0; t < 8; ++t) {
    v8f a0, a1;
#pragma unroll
    for (int r = 0; r < 8; ++r) { a0[r] = b0[r]; a1[r] = b1[r]; }
    for (int kb = 0; kb < 32; ++kb) {
      const _Float16* srow = state + (size_t)(n * PPTS + t * 16 + ln) * STCH + kb * 32 + hi * 16;
      H16 B; B.h[0] = *(const v8h*)srow; B.h[1] = *(const v8h*)(srow + 8);
      const _Float16* w0 = wf + (size_t)(m0 + ln) * STCH + kb * 32;
      const _Float16* w1 = wf + (size_t)(m0 + 16 + ln) * STCH + kb * 32;
      H16 A0; A0.h[0] = *(const v8h*)(w0 + hi * 16); A0.h[1] = *(const v8h*)(w0 + hi * 16 + 8);
      H16 A1; A1.h[0] = *(const v8h*)(w1 + hi * 16); A1.h[1] = *(const v8h*)(w1 + hi * 16 + 8);
      a0 = wmma_f16(A0.v, B.v, a0);
      a1 = wmma_f16(A1.v, B.v, a1);
    }
#pragma unroll
    for (int r = 0; r < 8; ++r) {
      rmax0[r] = fmaxf(rmax0[r], a0[r]);
      rmax1[r] = fmaxf(rmax1[r], a1[r]);
    }
  }
  // max across the 16 lanes holding the 16 columns of each tile
  for (int m = 8; m >= 1; m >>= 1) {
#pragma unroll
    for (int r = 0; r < 8; ++r) {
      rmax0[r] = fmaxf(rmax0[r], __shfl_xor(rmax0[r], m, 32));
      rmax1[r] = fmaxf(rmax1[r], __shfl_xor(rmax1[r], m, 32));
    }
  }
  if (ln == 0) {
#pragma unroll
    for (int r = 0; r < 8; ++r) {
      g[(size_t)n * FUS + m0 + hi * 8 + r]      = rmax0[r];
      g[(size_t)n * FUS + m0 + 16 + hi * 8 + r] = rmax1[r];
    }
  }
}

// ---------------------------------------------------------------------------
// Per-polygon bias vector: v[n][m] = b1[m] + W1[m, 0:256] . g[n]
// ---------------------------------------------------------------------------
__global__ __launch_bounds__(256)
void k_gvec(const float* __restrict__ g,
            const float* __restrict__ w_pred, const float* __restrict__ b_pred,
            const float* __restrict__ w_cls,  const float* __restrict__ b_cls,
            float* __restrict__ v_pred, float* __restrict__ v_cls) {
  int n = blockIdx.x, m = threadIdx.x;
  const float* w = blockIdx.y ? w_cls : w_pred;
  const float* b = blockIdx.y ? b_cls : b_pred;
  float* v = blockIdx.y ? v_cls : v_pred;
  const float* gg = g + (size_t)n * FUS;
  const float* wr = w + (size_t)m * (FUS + STCH);
  float s = b[m];
  for (int c = 0; c < FUS; ++c) s += wr[c] * gg[c];
  v[(size_t)n * HD1 + m] = s;
}

// ---------------------------------------------------------------------------
// Head layer 1: h1 = relu(W1[:,256:1280] @ state + v)  (pred / cls via grid.y)
// ---------------------------------------------------------------------------
__global__ __launch_bounds__(256)
void k_head1(const _Float16* __restrict__ state,
             const _Float16* __restrict__ w_pred, const _Float16* __restrict__ w_cls,
             const float* __restrict__ v_pred, const float* __restrict__ v_cls,
             _Float16* __restrict__ h1_pred, _Float16* __restrict__ h1_cls) {
  int n = blockIdx.x;
  const _Float16* w = blockIdx.y ? w_cls : w_pred;
  const float*    v = blockIdx.y ? v_cls : v_pred;
  _Float16*     out = blockIdx.y ? h1_cls : h1_pred;
  int tid = threadIdx.x;
  int wave = tid >> 5, lane = tid & 31, ln = lane & 15, hi = lane >> 4;
  int m0 = wave * 32;
  float c0[8], c1[8];
#pragma unroll
  for (int r = 0; r < 8; ++r) {
    c0[r] = v[(size_t)n * HD1 + m0 + hi * 8 + r];
    c1[r] = v[(size_t)n * HD1 + m0 + 16 + hi * 8 + r];
  }
  for (int t = 0; t < 8; ++t) {
    v8f a0, a1;
#pragma unroll
    for (int r = 0; r < 8; ++r) { a0[r] = c0[r]; a1[r] = c1[r]; }
    for (int kb = 0; kb < 32; ++kb) {
      const _Float16* srow = state + (size_t)(n * PPTS + t * 16 + ln) * STCH + kb * 32 + hi * 16;
      H16 B; B.h[0] = *(const v8h*)srow; B.h[1] = *(const v8h*)(srow + 8);
      const _Float16* w0 = w + (size_t)(m0 + ln) * (FUS + STCH) + FUS + kb * 32;
      const _Float16* w1 = w + (size_t)(m0 + 16 + ln) * (FUS + STCH) + FUS + kb * 32;
      H16 A0; A0.h[0] = *(const v8h*)(w0 + hi * 16); A0.h[1] = *(const v8h*)(w0 + hi * 16 + 8);
      H16 A1; A1.h[0] = *(const v8h*)(w1 + hi * 16); A1.h[1] = *(const v8h*)(w1 + hi * 16 + 8);
      a0 = wmma_f16(A0.v, B.v, a0);
      a1 = wmma_f16(A1.v, B.v, a1);
    }
    int p = t * 16 + ln;
    v8h r0, r1;
#pragma unroll
    for (int r = 0; r < 8; ++r) {
      float x0 = a0[r]; r0[r] = (_Float16)(x0 > 0.f ? x0 : 0.f);
      float x1 = a1[r]; r1[r] = (_Float16)(x1 > 0.f ? x1 : 0.f);
    }
    *(v8h*)&out[(size_t)(n * PPTS + p) * HD1 + m0 + hi * 8]      = r0;
    *(v8h*)&out[(size_t)(n * PPTS + p) * HD1 + m0 + 16 + hi * 8] = r1;
  }
}

// ---------------------------------------------------------------------------
// Head layer 2: h2 = relu(W2[64x256] @ h1 + b2)
// ---------------------------------------------------------------------------
__global__ __launch_bounds__(256)
void k_head2(const _Float16* __restrict__ h1_pred, const _Float16* __restrict__ h1_cls,
             const _Float16* __restrict__ w2_pred, const _Float16* __restrict__ w2_cls,
             const float* __restrict__ b2_pred, const float* __restrict__ b2_cls,
             _Float16* __restrict__ h2_pred, _Float16* __restrict__ h2_cls) {
  int n = blockIdx.x;
  const _Float16* h1 = blockIdx.y ? h1_cls : h1_pred;
  const _Float16* w2 = blockIdx.y ? w2_cls : w2_pred;
  const float*    b2 = blockIdx.y ? b2_cls : b2_pred;
  _Float16*       h2 = blockIdx.y ? h2_cls : h2_pred;
  int tid = threadIdx.x;
  int wave = tid >> 5, lane = tid & 31, ln = lane & 15, hi = lane >> 4;
  int m0 = (wave & 3) * 16;                 // 64 out rows over 4 row-tiles
  int cg = wave >> 2;                       // two col groups of 4 tiles each
  v8f acc[4];
#pragma unroll
  for (int tt = 0; tt < 4; ++tt)
#pragma unroll
    for (int r = 0; r < 8; ++r) acc[tt][r] = b2[m0 + hi * 8 + r];
  for (int kb = 0; kb < 8; ++kb) {
    const _Float16* w0 = w2 + (size_t)(m0 + ln) * HD1 + kb * 32;
    H16 A; A.h[0] = *(const v8h*)(w0 + hi * 16); A.h[1] = *(const v8h*)(w0 + hi * 16 + 8);
#pragma unroll
    for (int tt = 0; tt < 4; ++tt) {
      int t = cg * 4 + tt;
      const _Float16* br = h1 + (size_t)(n * PPTS + t * 16 + ln) * HD1 + kb * 32 + hi * 16;
      H16 B; B.h[0] = *(const v8h*)br; B.h[1] = *(const v8h*)(br + 8);
      acc[tt] = wmma_f16(A.v, B.v, acc[tt]);
    }
  }
#pragma unroll
  for (int tt = 0; tt < 4; ++tt) {
    int p = (cg * 4 + tt) * 16 + ln;
    v8h res;
#pragma unroll
    for (int r = 0; r < 8; ++r) {
      float x = acc[tt][r];
      res[r] = (_Float16)(x > 0.f ? x : 0.f);
    }
    *(v8h*)&h2[(size_t)(n * PPTS + p) * HD2 + m0 + hi * 8] = res;
  }
}

// ---------------------------------------------------------------------------
// Final tiny matvecs + epilogue: i_poly, cls, sigmoid + circular NMS
// ---------------------------------------------------------------------------
__global__ __launch_bounds__(128)
void k_final(const _Float16* __restrict__ h2p, const _Float16* __restrict__ h2c,
             const float* __restrict__ w3p, const float* __restrict__ b3p,
             const float* __restrict__ w3c, const float* __restrict__ b3c,
             const float* __restrict__ ipoly, float* __restrict__ out) {
  __shared__ float sig[PPTS];
  int n = blockIdx.x, p = threadIdx.x;
  size_t q = (size_t)n * PPTS + p;
  const _Float16* hp = h2p + q * HD2;
  const _Float16* hc = h2c + q * HD2;
  float ox = b3p[0], oy = b3p[1], cv = b3c[0];
  for (int c = 0; c < HD2; ++c) {
    float hv = (float)hp[c];
    ox += w3p[c] * hv;
    oy += w3p[HD2 + c] * hv;
    cv += w3c[c] * (float)hc[c];
  }
  out[q * 2 + 0] = ipoly[q * 2 + 0] * 4.0f + ox;
  out[q * 2 + 1] = ipoly[q * 2 + 1] * 4.0f + oy;
  out[(size_t)NPOLY * PPTS * 2 + q] = cv;
  float s = 1.0f / (1.0f + __expf(-cv));
  sig[p] = s;
  __syncthreads();
  float m = s;
#pragma unroll
  for (int d = -2; d <= 2; ++d) m = fmaxf(m, sig[(p + d) & 127]);
  out[(size_t)NPOLY * PPTS * 3 + q] = (s >= m) ? s : 0.0f;
}

// ---------------------------------------------------------------------------
// Host launcher
// ---------------------------------------------------------------------------
extern "C" void kernel_launch(void* const* d_in, const int* in_sizes, int n_in,
                              void* d_out, int out_size, void* d_ws, size_t ws_size,
                              hipStream_t stream) {
  (void)in_sizes; (void)n_in; (void)out_size; (void)ws_size;
  const float* cnn      = (const float*)d_in[0];
  const float* ipoly    = (const float*)d_in[1];
  const float* cpoly    = (const float*)d_in[2];
  const float* itcls    = (const float*)d_in[3];
  const int*   ind      = (const int*)d_in[4];
  const float* head_w   = (const float*)d_in[5];
  const float* head_b   = (const float*)d_in[6];
  const float* res_w    = (const float*)d_in[7];
  const float* res_b    = (const float*)d_in[8];
  const float* fusion_w = (const float*)d_in[9];
  const float* fusion_b = (const float*)d_in[10];
  const float* pred1_w  = (const float*)d_in[11];
  const float* pred1_b  = (const float*)d_in[12];
  const float* pred2_w  = (const float*)d_in[13];
  const float* pred2_b  = (const float*)d_in[14];
  const float* pred3_w  = (const float*)d_in[15];
  const float* pred3_b  = (const float*)d_in[16];
  const float* cls1_w   = (const float*)d_in[17];
  const float* cls1_b   = (const float*)d_in[18];
  const float* cls2_w   = (const float*)d_in[19];
  const float* cls2_b   = (const float*)d_in[20];
  const float* cls3_w   = (const float*)d_in[21];
  const float* cls3_b   = (const float*)d_in[22];
  float* out = (float*)d_out;

  char* ws = (char*)d_ws;
  size_t off = 0;
  auto take = [&](size_t bytes) -> char* {
    char* p = ws + off;
    off += bytes;
    off = (off + 255) & ~(size_t)255;
    return p;
  };
  _Float16* xpad  = (_Float16*)take((size_t)NPOLY * PPTS * FDPAD * 2);
  _Float16* state = (_Float16*)take((size_t)NPOLY * PPTS * STCH * 2);
  float*    g     = (float*)   take((size_t)NPOLY * FUS * 4);
  float*    vpred = (float*)   take((size_t)NPOLY * HD1 * 4);
  float*    vcls  = (float*)   take((size_t)NPOLY * HD1 * 4);
  _Float16* h1p   = (_Float16*)take((size_t)NPOLY * PPTS * HD1 * 2);
  _Float16* h1c   = (_Float16*)take((size_t)NPOLY * PPTS * HD1 * 2);
  _Float16* h2p   = (_Float16*)take((size_t)NPOLY * PPTS * HD2 * 2);
  _Float16* h2c   = (_Float16*)take((size_t)NPOLY * PPTS * HD2 * 2);
  _Float16* whead = (_Float16*)take((size_t)KTAP * SD * FDPAD * 2);
  _Float16* wres  = (_Float16*)take((size_t)7 * KTAP * SD * SD * 2);
  _Float16* wfus  = (_Float16*)take((size_t)FUS * STCH * 2);
  _Float16* w1p   = (_Float16*)take((size_t)HD1 * (FUS + STCH) * 2);
  _Float16* w1c   = (_Float16*)take((size_t)HD1 * (FUS + STCH) * 2);
  _Float16* w2p   = (_Float16*)take((size_t)HD2 * HD1 * 2);
  _Float16* w2c   = (_Float16*)take((size_t)HD2 * HD1 * 2);

  // 1. weight conversion
  k_cvt_head<<<(KTAP * SD * FDPAD + 255) / 256, 256, 0, stream>>>(head_w, whead);
  k_cvt_res<<<(7 * KTAP * SD * SD + 255) / 256, 256, 0, stream>>>(res_w, wres);
  k_cvt_copy<<<256, 256, 0, stream>>>(fusion_w, wfus, FUS * STCH);
  k_cvt_copy<<<256, 256, 0, stream>>>(pred1_w, w1p, HD1 * (FUS + STCH));
  k_cvt_copy<<<256, 256, 0, stream>>>(cls1_w, w1c, HD1 * (FUS + STCH));
  k_cvt_copy<<<64, 256, 0, stream>>>(pred2_w, w2p, HD2 * HD1);
  k_cvt_copy<<<64, 256, 0, stream>>>(cls2_w, w2c, HD2 * HD1);

  // 2. bilinear gather + input assembly
  k_gather<<<NPOLY, 128, 0, stream>>>(cnn, ipoly, cpoly, itcls, ind, xpad);

  // 3. head conv (67->128, d=1) into state slice 0
  k_conv<<<NPOLY, 256, 0, stream>>>(xpad, FDPAD, FDPAD, whead, head_b,
                                    state + 0, 1, 0);

  // 4. seven residual dilated circular convs
  const int dils[7] = {1, 1, 1, 2, 2, 4, 4};
  for (int l = 0; l < 7; ++l) {
    k_conv<<<NPOLY, 256, 0, stream>>>(state + l * SD, STCH, SD,
                                      wres + (size_t)l * KTAP * SD * SD,
                                      res_b + l * SD,
                                      state + (l + 1) * SD, dils[l], 1);
  }

  // 5. fusion 1x1 + global max
  k_fusion<<<NPOLY, 256, 0, stream>>>(state, wfus, fusion_b, g);

  // 6. fold g into per-polygon head bias vectors
  k_gvec<<<dim3(NPOLY, 2), 256, 0, stream>>>(g, pred1_w, pred1_b, cls1_w, cls1_b,
                                             vpred, vcls);

  // 7. head layers
  k_head1<<<dim3(NPOLY, 2), 256, 0, stream>>>(state, w1p, w1c, vpred, vcls, h1p, h1c);
  k_head2<<<dim3(NPOLY, 2), 256, 0, stream>>>(h1p, h1c, w2p, w2c, pred2_b, cls2_b,
                                              h2p, h2c);

  // 8. final matvecs + epilogue + NMS
  k_final<<<NPOLY, 128, 0, stream>>>(h2p, h2c, pred3_w, pred3_b, cls3_w, cls3_b,
                                     ipoly, out);
}
